// FC_Decoder_4432406249542
// MI455X (gfx1250) — matrix operand order repaired and checked by reading, not compiled, not run on hardware
//
#include <hip/hip_runtime.h>
#include <hip/hip_bf16.h>

// ---------------- dims ----------------
#define D_MODEL 512
#define D_INNER 1024
#define D_STATE 16
#define D_CONV 4
#define DT_RANK 32
#define N_CLASS 1500
#define BM 3
#define SEQ_L 8192
#define TOKENS (BM * SEQ_L)      // 24576
#define H1 (D_MODEL * BM)        // 1536

// GEMM tiling
#define KSTEP   32               // WMMA K per step
#define KSTEP_P 36               // padded LDS row stride (floats) to spread banks
#define TILE_N  64               // block-shared W tile columns
#define TILE_M  256              // 8 waves * 32 rows

typedef __attribute__((ext_vector_type(16))) __bf16 v16bf;
typedef __attribute__((ext_vector_type(8)))  float  v8f;

// ---- CDNA5 async global->LDS copy (ASYNCcnt-tracked DMA), 16 bytes ----
__device__ __forceinline__ void async_copy_b128(unsigned lds_addr, const float* g) {
    asm volatile("global_load_async_to_lds_b128 %0, %1, off"
                 :: "v"(lds_addr), "v"((unsigned long long)(uintptr_t)g)
                 : "memory");
}

// ============================================================
// Generic GEMM: C[M,N] = act( A[M,K](lda) @ W[N,K]^T + bias )
// Block = 8 waves -> 256x64 C tile. The shared 64xKSTEP W tile
// is DMA'd into LDS with global_load_async_to_lds_b128 and
// double-buffered (prefetch overlapped with WMMA; s_wait_asynccnt
// + barrier for handoff). Each wave: 2 M-subtiles x 4 N-subtiles
// = 8 v_wmma_f32_16x16x32_bf16 per K-step; A fragments from
// global (rows private per wave), B fragments from LDS.
// act: 0=none 1=relu 2=sigmoid 3=softplus
// ============================================================
template <bool EDGE>
__global__ void gemm_wmma_bf16(const float* __restrict__ A,
                               const float* __restrict__ W,
                               const float* __restrict__ bias,
                               float* __restrict__ C,
                               int M, int N, int K, int lda,
                               int tilesN, int act) {
    __shared__ float sbuf[2][TILE_N * KSTEP_P];      // 2 x 9 KB of 320 KB LDS

    const int tid  = threadIdx.x;
    const int lane = tid & 31;
    const int wave = tid >> 5;
    const int tN   = blockIdx.x % tilesN;
    const int mb   = blockIdx.x / tilesN;

    const int  l15 = lane & 15;
    const bool hi  = (lane >= 16);
    const int rowA0 = mb * TILE_M + wave * 32 + l15;  // A layout: M = lane&15
    const int rowA1 = rowA0 + 16;

    // ---- staging map: 512 16B segments (64 cols x 8), 2 per thread ----
    const int sid0 = tid,        col0 = sid0 >> 3, seg0 = sid0 & 7;
    const int sid1 = tid + 256,  col1 = sid1 >> 3, seg1 = sid1 & 7;
    int cg0 = tN * TILE_N + col0;
    int cg1 = tN * TILE_N + col1;
    if (EDGE) {                                       // clamp: garbage cols never stored
        cg0 = (cg0 < N) ? cg0 : (N - 1);
        cg1 = (cg1 < N) ? cg1 : (N - 1);
    }
    const float* gW0 = W + (size_t)cg0 * K + seg0 * 4;
    const float* gW1 = W + (size_t)cg1 * K + seg1 * 4;
    unsigned lds0[2], lds1[2];
#pragma unroll
    for (int b = 0; b < 2; ++b) {
        lds0[b] = (unsigned)(uintptr_t)&sbuf[b][col0 * KSTEP_P + seg0 * 4];
        lds1[b] = (unsigned)(uintptr_t)&sbuf[b][col1 * KSTEP_P + seg1 * 4];
    }

    v8f acc[2][4];
#pragma unroll
    for (int mi = 0; mi < 2; ++mi)
#pragma unroll
        for (int s = 0; s < 4; ++s) acc[mi][s] = (v8f){};

    // ---- prologue: DMA first W tile ----
    async_copy_b128(lds0[0], gW0);
    async_copy_b128(lds1[0], gW1);

    int cur = 0;
    for (int k0 = 0; k0 < K; k0 += KSTEP) {
        const bool last = (k0 + KSTEP >= K);          // wave-uniform branch
        if (!last) {                                  // prefetch next tile
            async_copy_b128(lds0[cur ^ 1], gW0 + k0 + KSTEP);
            async_copy_b128(lds1[cur ^ 1], gW1 + k0 + KSTEP);
            asm volatile("s_wait_asynccnt 2" ::: "memory");  // current tile landed
        } else {
            asm volatile("s_wait_asynccnt 0" ::: "memory");
        }
        __syncthreads();                              // tile visible to all waves

        // ---- two A fragments (16x32 bf16), rows private to this wave ----
        v16bf a0, a1;
        {
            const float* ap0 = A + (size_t)rowA0 * lda + k0 + (hi ? 8 : 0);
            const float* ap1 = A + (size_t)rowA1 * lda + k0 + (hi ? 8 : 0);
#pragma unroll
            for (int j = 0; j < 8; ++j) {
                a0[j]     = (__bf16)ap0[j];
                a0[8 + j] = (__bf16)ap0[16 + j];
                a1[j]     = (__bf16)ap1[j];
                a1[8 + j] = (__bf16)ap1[16 + j];
            }
        }
        // ---- four B fragments (32x16 bf16) from the shared LDS tile ----
        const float* sb = sbuf[cur];
#pragma unroll
        for (int s = 0; s < 4; ++s) {
            v16bf b;
            const float* bp = sb + (s * 16 + l15) * KSTEP_P + (hi ? 8 : 0);
#pragma unroll
            for (int j = 0; j < 8; ++j) {
                b[j]     = (__bf16)bp[j];
                b[8 + j] = (__bf16)bp[16 + j];
            }
            acc[0][s] = __builtin_amdgcn_wmma_f32_16x16x32_bf16(
                false, a0, false, b, (short)0, acc[0][s], false, false);
            acc[1][s] = __builtin_amdgcn_wmma_f32_16x16x32_bf16(
                false, a1, false, b, (short)0, acc[1][s], false, false);
        }
        __syncthreads();                              // all reads done before overwrite
        cur ^= 1;
    }

    // ---- epilogue: bias + activation + store ----
#pragma unroll
    for (int s = 0; s < 4; ++s) {
        const int col = tN * TILE_N + s * 16 + l15;
        if (!EDGE || col < N) {                       // divergent store guard only
            float bv = bias ? bias[col] : 0.0f;
#pragma unroll
            for (int mi = 0; mi < 2; ++mi) {
#pragma unroll
                for (int i = 0; i < 8; ++i) {
                    int r = mb * TILE_M + wave * 32 + mi * 16 + i + (hi ? 8 : 0);
                    float v = acc[mi][s][i] + bv;
                    if      (act == 1) v = fmaxf(v, 0.0f);
                    else if (act == 2) v = 1.0f / (1.0f + __expf(-v));
                    else if (act == 3) v = (v > 20.0f) ? v : __logf(1.0f + __expf(v));
                    C[(size_t)r * N + col] = v;
                }
            }
        }
    }
}

// ============================================================
// Depthwise causal conv1d (kernel 4, left pad 3) + bias + SiLU
// ============================================================
__global__ void conv_silu_kernel(const float* __restrict__ xz,
                                 const float* __restrict__ cw,
                                 const float* __restrict__ cb,
                                 float* __restrict__ xc) {
    size_t idx = (size_t)blockIdx.x * blockDim.x + threadIdx.x;
    if (idx >= (size_t)TOKENS * D_INNER) return;
    int d = idx % D_INNER;
    size_t tok = idx / D_INNER;          // b*L + l
    int l = tok % SEQ_L;
    size_t brow = tok - l;               // b*L
    float s = cb[d];
#pragma unroll
    for (int k = 0; k < D_CONV; ++k) {
        int ll = l - (D_CONV - 1) + k;
        if (ll >= 0)
            s += cw[d * D_CONV + k] * xz[(brow + ll) * (2 * D_INNER) + d];
    }
    xc[idx] = s / (1.0f + __expf(-s));   // silu
}

// ============================================================
// Selective scan. 16 lanes = 16 states of one (b,d) channel.
// Fused: y = (scan_y + x*D) * silu(z)
// ============================================================
__global__ void scan_kernel(const float* __restrict__ delta,
                            const float* __restrict__ xdbl,   // (TOKENS, 64): [dt|B|C]
                            const float* __restrict__ xc,
                            const float* __restrict__ xz,     // for z
                            const float* __restrict__ A_log,
                            const float* __restrict__ D_param,
                            float* __restrict__ y) {
    int grp = blockIdx.x * (blockDim.x >> 4) + (threadIdx.x >> 4); // (b,d) pair
    int n = threadIdx.x & 15;
    int b = grp / D_INNER;
    int d = grp - b * D_INNER;
    if (b >= BM) return;

    const float Av = -__expf(A_log[d * D_STATE + n]);
    const float Dv = D_param[d];
    float h = 0.0f;
    const size_t base = (size_t)b * SEQ_L;

    for (int l = 0; l < SEQ_L; ++l) {
        size_t tok = base + l;
        float dl = delta[tok * D_INNER + d];
        float xt = xc[tok * D_INNER + d];
        float Bt = xdbl[tok * 64 + DT_RANK + n];
        float Ct = xdbl[tok * 64 + DT_RANK + D_STATE + n];
        h = h * __expf(dl * Av) + dl * xt * Bt;
        float yv = h * Ct;
        yv += __shfl_xor(yv, 1, 32);
        yv += __shfl_xor(yv, 2, 32);
        yv += __shfl_xor(yv, 4, 32);
        yv += __shfl_xor(yv, 8, 32);
        if (n == 0) {
            float zt = xz[tok * (2 * D_INNER) + D_INNER + d];
            float sz = zt / (1.0f + __expf(-zt));
            y[tok * D_INNER + d] = (yv + xt * Dv) * sz;
        }
    }
}

// ============================================================
// (BM, L, D_MODEL) -> (L, BM*D_MODEL) transpose for the MLP head
// ============================================================
__global__ void head_transpose_kernel(const float* __restrict__ mo,
                                      float* __restrict__ mi) {
    size_t idx = (size_t)blockIdx.x * blockDim.x + threadIdx.x;
    if (idx >= (size_t)SEQ_L * H1) return;
    int c = idx % H1;          // b*512 + d
    int l = idx / H1;
    int b = c / D_MODEL;
    int d = c - b * D_MODEL;
    mi[idx] = mo[((size_t)b * SEQ_L + l) * D_MODEL + d];
}

// ---------------- host-side launch helper ----------------
static inline void launch_gemm(const float* A, const float* W, const float* bias,
                               float* C, int M, int N, int K, int lda, int act,
                               hipStream_t stream) {
    int tilesN = (N + TILE_N - 1) / TILE_N;
    int blocks = (M / TILE_M) * tilesN;        // M is a multiple of 256 for all calls
    if (N % TILE_N == 0)
        gemm_wmma_bf16<false><<<blocks, 256, 0, stream>>>(A, W, bias, C, M, N, K, lda, tilesN, act);
    else
        gemm_wmma_bf16<true ><<<blocks, 256, 0, stream>>>(A, W, bias, C, M, N, K, lda, tilesN, act);
}

extern "C" void kernel_launch(void* const* d_in, const int* in_sizes, int n_in,
                              void* d_out, int out_size, void* d_ws, size_t ws_size,
                              hipStream_t stream) {
    (void)in_sizes; (void)n_in; (void)out_size; (void)ws_size;
    const float* hs        = (const float*)d_in[0];
    const float* in_proj_w = (const float*)d_in[1];
    const float* conv_w    = (const float*)d_in[2];
    const float* conv_b    = (const float*)d_in[3];
    const float* x_proj_w  = (const float*)d_in[4];
    const float* dt_proj_w = (const float*)d_in[5];
    const float* dt_proj_b = (const float*)d_in[6];
    const float* A_log     = (const float*)d_in[7];
    const float* D_param   = (const float*)d_in[8];
    const float* out_proj_w= (const float*)d_in[9];
    const float* w1 = (const float*)d_in[10]; const float* b1 = (const float*)d_in[11];
    const float* w2 = (const float*)d_in[12]; const float* b2 = (const float*)d_in[13];
    const float* w3 = (const float*)d_in[14]; const float* b3 = (const float*)d_in[15];
    float* out = (float*)d_out;

    // workspace layout (fp32 elements)
    float* ws = (float*)d_ws;
    float* xz     = ws;                              // TOKENS * 2048
    float* xc     = xz    + (size_t)TOKENS * 2048;   // TOKENS * 1024
    float* xdbl   = xc    + (size_t)TOKENS * 1024;   // TOKENS * 64
    float* delta  = xdbl  + (size_t)TOKENS * 64;     // TOKENS * 1024
    float* yb     = delta + (size_t)TOKENS * 1024;   // TOKENS * 1024
    float* mo     = yb    + (size_t)TOKENS * 1024;   // TOKENS * 512
    float* mi     = mo    + (size_t)TOKENS * 512;    // SEQ_L * 1536
    float* h1     = mi    + (size_t)SEQ_L * H1;      // SEQ_L * 512
    float* h2     = h1    + (size_t)SEQ_L * 512;     // SEQ_L * 512

    // 1) in_proj: xz = hs @ in_proj_w^T        (24576 x 2048, K=512)
    launch_gemm(hs, in_proj_w, nullptr, xz, TOKENS, 2 * D_INNER, D_MODEL, D_MODEL, 0, stream);

    // 2) depthwise causal conv + silu
    {
        size_t n = (size_t)TOKENS * D_INNER;
        conv_silu_kernel<<<(unsigned)((n + 255) / 256), 256, 0, stream>>>(xz, conv_w, conv_b, xc);
    }

    // 3) x_proj: xdbl = xc @ x_proj_w^T        (24576 x 64, K=1024)
    launch_gemm(xc, x_proj_w, nullptr, xdbl, TOKENS, DT_RANK + 2 * D_STATE, D_INNER, D_INNER, 0, stream);

    // 4) dt_proj + bias + softplus: delta = softplus(xdbl[:, :32] @ dt_proj_w^T + b)
    launch_gemm(xdbl, dt_proj_w, dt_proj_b, delta, TOKENS, D_INNER, DT_RANK, 64, 3, stream);

    // 5) selective scan (fused +x*D and *silu(z))
    {
        int groups = BM * D_INNER;                 // 3072 (b,d) channels
        int blocks = groups / 16;                  // 192 blocks of 256 threads
        scan_kernel<<<blocks, 256, 0, stream>>>(delta, xdbl, xc, xz, A_log, D_param, yb);
    }

    // 6) out_proj: mo = y @ out_proj_w^T        (24576 x 512, K=1024)
    launch_gemm(yb, out_proj_w, nullptr, mo, TOKENS, D_MODEL, D_INNER, D_INNER, 0, stream);

    // 7) (BM,L,512) -> (L,1536)
    {
        size_t n = (size_t)SEQ_L * H1;
        head_transpose_kernel<<<(unsigned)((n + 255) / 256), 256, 0, stream>>>(mo, mi);
    }

    // 8) MLP head
    launch_gemm(mi, w1, b1, h1, SEQ_L, D_MODEL, H1, H1, 1, stream);          // relu
    launch_gemm(h1, w2, b2, h2, SEQ_L, D_MODEL, D_MODEL, D_MODEL, 2, stream); // sigmoid
    launch_gemm(h2, w3, b3, out, SEQ_L, N_CLASS, D_MODEL, D_MODEL, 0, stream);
}
